// DeepRIRNet_1168231105196
// MI455X (gfx1250) — matrix operand order repaired and checked by reading, not compiled, not run on hardware
//
#include <hip/hip_runtime.h>
#include <hip/hip_bf16.h>
#include <stdint.h>
#include <stddef.h>

// ---------------- problem constants ----------------
static constexpr int kB   = 32;
static constexpr int kT   = 2048;
static constexpr int kH   = 512;
static constexpr int kG4  = 4 * kH;       // 2048 gate rows
static constexpr int kL   = 6;
static constexpr int kNWG = 16;           // one cluster: 16 WGs, one per WGP
static constexpr int kNTH = 256;          // 8 waves of 32
static constexpr int kJPW = kH / kNWG;    // 32 hidden units per WG
static constexpr float kEps = 1e-5f;

typedef __bf16 bf16_t;
typedef __attribute__((ext_vector_type(8)))  bf16_t v8bf;
typedef __attribute__((ext_vector_type(16))) bf16_t v16bf;
typedef __attribute__((ext_vector_type(8)))  float  v8f;
typedef __attribute__((ext_vector_type(4)))  unsigned int v4u;
typedef __attribute__((ext_vector_type(8)))  int v8i;
typedef __attribute__((ext_vector_type(4)))  int v4i;

// ---------------- workspace layout ----------------
static constexpr size_t kAlign = 256;
static constexpr size_t align_up(size_t x) { return (x + kAlign - 1) & ~(kAlign - 1); }

static constexpr size_t kHbufBytes   = (size_t)kB * kT * kH * 4;  // fp32 layer io (residual stream)
static constexpr size_t kOutbufBytes = (size_t)kB * kT * kH * 4;  // fp32 raw LSTM output
static constexpr size_t kShadowBytes = (size_t)kB * kT * kH * 2;  // bf16 shadow of layer input
static constexpr size_t kWihbBytes   = (size_t)kL * kG4 * kH * 2; // bf16 W_ih
static constexpr size_t kWhhbBytes   = (size_t)kL * kG4 * kH * 2; // bf16 W_hh
static constexpr size_t kXpBytes     = (size_t)kB * kH * 4;       // x projection
static constexpr size_t kHstateBytes = (size_t)kB * kH * 2;       // bf16 recurrent h state

static constexpr size_t kOffHbuf   = 0;
static constexpr size_t kOffOutbuf = align_up(kOffHbuf + kHbufBytes);
static constexpr size_t kOffShadow = align_up(kOffOutbuf + kOutbufBytes);
static constexpr size_t kOffWihb   = align_up(kOffShadow + kShadowBytes);
static constexpr size_t kOffWhhb   = align_up(kOffWihb + kWihbBytes);
static constexpr size_t kOffXp     = align_up(kOffWhhb + kWhhbBytes);
static constexpr size_t kOffHstate = align_up(kOffXp + kXpBytes);
static constexpr size_t kOffBar    = align_up(kOffHstate + kHstateBytes);

// ---------------- LDS layout (dynamic shared) of recurrent kernel ----------------
static constexpr unsigned kLdsWih    = 0;                              // 128 rows x 512 bf16 = 128 KB
static constexpr unsigned kLdsWhh    = kLdsWih + 128 * kH * 2;         // +128 KB
static constexpr unsigned kLdsHprev  = kLdsWhh + 128 * kH * 2;         // +32 KB (full h state, 32x512 bf16)
static constexpr unsigned kLdsGbuf   = kLdsHprev + kB * kH * 2;        // +16 KB (4 gates x 32 u x 32 b f32)
static constexpr unsigned kLdsCstate = kLdsGbuf + 4 * kJPW * kB * 4;   // +4 KB
static constexpr unsigned kLdsBias   = kLdsCstate + kJPW * kB * 4;     // +512 B
static constexpr unsigned kLdsTotal  = kLdsBias + 4 * kJPW * 4;        // 315904 B <= 320 KB
static_assert(kLdsTotal <= 320 * 1024, "LDS budget");

// ---------------- device helpers ----------------
__device__ __forceinline__ v8f wmma_bf16(v16bf a, v16bf b, v8f c) {
  return __builtin_amdgcn_wmma_f32_16x16x32_bf16(false, a, false, b, (short)0, c, false, false);
}

__device__ __forceinline__ v16bf cat8(v8bf lo, v8bf hi) {
  return __builtin_shufflevector(lo, hi, 0, 1, 2, 3, 4, 5, 6, 7,
                                         8, 9, 10, 11, 12, 13, 14, 15);
}

// TDM descriptor (D#) for a 2D tile of 2-byte elements: tile_d0 x tile_d1,
// line stride stride0 (elements). Groups packed per CDNA5 ISA 8.3/8.4.
// This toolchain's builtin takes 6 args: (g0, g1, g2, g3, g4, cpol).
__device__ __forceinline__ void tdm_load_2d(unsigned lds_byte_off, const void* gsrc,
                                            unsigned tensor_d0, unsigned tensor_d1,
                                            unsigned tile_d0, unsigned tile_d1,
                                            unsigned long long stride0_elems) {
  unsigned long long ga = (unsigned long long)(uintptr_t)gsrc;
  v4u g0 = {
      1u,                                                  // count=1, user descriptor
      lds_byte_off,                                        // lds_addr (bytes)
      (unsigned)(ga & 0xFFFFFFFFu),                        // global_addr[31:0]
      (unsigned)((ga >> 32) & 0x01FFFFFFu) | (2u << 30)    // global_addr[56:32] | type=2
  };
  v8i g1;
  g1[0] = (int)(1u << 16);                                           // data_size=2B, mask=0
  g1[1] = (int)((tensor_d0 & 0xFFFFu) << 16);                        // tensor_dim0[15:0]
  g1[2] = (int)(((tensor_d0 >> 16) & 0xFFFFu) | ((tensor_d1 & 0xFFFFu) << 16));
  g1[3] = (int)(((tensor_d1 >> 16) & 0xFFFFu) | ((tile_d0 & 0xFFFFu) << 16));
  g1[4] = (int)(tile_d1 & 0xFFFFu);                                  // tile_dim1 (tile_dim2=0)
  g1[5] = (int)(stride0_elems & 0xFFFFFFFFull);                      // dim0_stride[31:0]
  g1[6] = (int)((stride0_elems >> 32) & 0xFFFFull);                  // dim0_stride[47:32]
  g1[7] = 0;
  v4i gz  = {0, 0, 0, 0};
  v8i gz8 = {0, 0, 0, 0, 0, 0, 0, 0};
  __builtin_amdgcn_tensor_load_to_lds(g0, g1, gz, gz, gz8, 0);
}

// Cluster-wide sync: s_cluster_barrier (real sync under cluster dispatch; NOP otherwise)
// plus a sense-reversing atomic barrier that is correct for any co-resident 16-WG grid.
__device__ __forceinline__ void grid_sync(unsigned* bar) {
  __threadfence();
  __syncthreads();
  __builtin_amdgcn_s_cluster_barrier();
  if (threadIdx.x == 0) {
    unsigned gen = __hip_atomic_load(bar + 1, __ATOMIC_ACQUIRE, __HIP_MEMORY_SCOPE_AGENT);
    unsigned arr = __hip_atomic_fetch_add(bar, 1u, __ATOMIC_ACQ_REL, __HIP_MEMORY_SCOPE_AGENT);
    if (arr == (unsigned)kNWG - 1u) {
      __hip_atomic_store(bar, 0u, __ATOMIC_RELAXED, __HIP_MEMORY_SCOPE_AGENT);
      __hip_atomic_fetch_add(bar + 1, 1u, __ATOMIC_ACQ_REL, __HIP_MEMORY_SCOPE_AGENT);
    } else {
      while (__hip_atomic_load(bar + 1, __ATOMIC_ACQUIRE, __HIP_MEMORY_SCOPE_AGENT) == gen) { }
    }
  }
  __syncthreads();
}

__device__ __forceinline__ float sigmoidf_(float x) { return 1.0f / (1.0f + __expf(-x)); }

// ---------------- kernels ----------------
__global__ void inproj_kernel(const float* __restrict__ x, const float* __restrict__ w,
                              const float* __restrict__ b, float* __restrict__ xp,
                              unsigned* __restrict__ bar) {
  int idx = blockIdx.x * blockDim.x + threadIdx.x;
  if (idx == 0) bar[0] = 0u;   // arm the inter-WG barrier counter for this launch
  if (idx < kB * kH) {
    int bb = idx / kH, hh = idx % kH;
    float acc = b[hh];
#pragma unroll
    for (int k = 0; k < 12; ++k) acc += x[bb * 12 + k] * w[hh * 12 + k];
    xp[idx] = acc;
  }
}

__global__ void bcast_kernel(const float* __restrict__ xp, float* __restrict__ hbuf,
                             bf16_t* __restrict__ shadow) {
  size_t gid = (size_t)blockIdx.x * blockDim.x + threadIdx.x;   // over B*T*H exactly
  size_t th = (size_t)kT * kH;
  int bb = (int)(gid / th);
  int hh = (int)(gid % (size_t)kH);
  float v = xp[bb * kH + hh];
  hbuf[gid] = v;
  shadow[gid] = (bf16_t)v;
}

__global__ void wconv_kernel(const float* __restrict__ wih, const float* __restrict__ whh,
                             bf16_t* __restrict__ wihb, bf16_t* __restrict__ whhb) {
  size_t n = (size_t)kL * kG4 * kH;
  for (size_t i = (size_t)blockIdx.x * blockDim.x + threadIdx.x; i < n;
       i += (size_t)gridDim.x * blockDim.x) {
    wihb[i] = (bf16_t)wih[i];
    whhb[i] = (bf16_t)whh[i];
  }
}

// Persistent fused LSTM layer: 16 WGs (one per WGP), W_ih/W_hh slices resident in LDS.
__global__ __launch_bounds__(kNTH)
void recur_kernel(const bf16_t* __restrict__ wihb, const bf16_t* __restrict__ whhb,
                  const float* __restrict__ bih, const float* __restrict__ bhh,
                  const bf16_t* __restrict__ hin,      // [B][T][H] bf16 layer input
                  float* __restrict__ outb,            // [B][T][H] f32 LSTM output
                  bf16_t* __restrict__ hstate,         // [B][H] bf16 recurrent state
                  unsigned* __restrict__ bar, int layer) {
  extern __shared__ char smem[];
  bf16_t* wih_s   = (bf16_t*)(smem + kLdsWih);
  bf16_t* whh_s   = (bf16_t*)(smem + kLdsWhh);
  bf16_t* hprev_s = (bf16_t*)(smem + kLdsHprev);
  float*  gbuf    = (float*)(smem + kLdsGbuf);     // [4*32 gate rows][32 batch]
  float*  cstate  = (float*)(smem + kLdsCstate);   // [32 units][32 batch]
  float*  bias_s  = (float*)(smem + kLdsBias);     // [4*32] (b_ih + b_hh)

  const int tid  = threadIdx.x;
  const int wg   = blockIdx.x;           // 0..15  (== WG-in-cluster id)
  const int j0   = wg * kJPW;            // first hidden unit owned by this WG
  const int wave = tid >> 5;
  const int lane = tid & 31;
  const int r16  = lane & 15;            // row (A) / column (B) within 16-wide tile
  const int kh   = lane >> 4;            // K-half selector per WMMA VGPR layout
  const int gate = wave >> 1;            // each pair of waves handles one gate type
  const int ublk = (wave & 1) * 16;      // 16-unit sub-slice within the 32 units
  const unsigned lrow0 = (unsigned)wave * 16;   // LDS weight-row base for this wave

  // ---- init: cell state, biases, own h-state slice, weight slices via TDM ----
  for (int i = tid; i < kJPW * kB; i += kNTH) cstate[i] = 0.0f;
  for (int i = tid; i < 4 * kJPW; i += kNTH) {
    int g = i / kJPW, u = i % kJPW;
    int grow = g * kH + j0 + u;
    bias_s[i] = bih[layer * kG4 + grow] + bhh[layer * kG4 + grow];
  }
  for (int i = tid; i < kB * kJPW; i += kNTH) {
    int b = i / kJPW, u = i % kJPW;
    hstate[(size_t)b * kH + j0 + u] = (bf16_t)0.0f;
  }
  if (wave == 0) {
    const bf16_t* wl_i = wihb + (size_t)layer * kG4 * kH;
    const bf16_t* wl_h = whhb + (size_t)layer * kG4 * kH;
#pragma unroll
    for (int g = 0; g < 4; ++g) {   // 32 rows x 512 cols per gate segment
      tdm_load_2d(kLdsWih + (unsigned)g * kJPW * kH * 2,
                  wl_i + (size_t)(g * kH + j0) * kH, kH, kJPW, kH, kJPW, kH);
      tdm_load_2d(kLdsWhh + (unsigned)g * kJPW * kH * 2,
                  wl_h + (size_t)(g * kH + j0) * kH, kH, kJPW, kH, kJPW, kH);
    }
    __builtin_amdgcn_s_wait_tensorcnt(0);
  }
  grid_sync(bar);

  const bf16_t* arow0 = hin + (size_t)r16 * kT * kH;         // batch rows 0..15
  const bf16_t* arow1 = hin + (size_t)(16 + r16) * kT * kH;  // batch rows 16..31

  for (int t = 0; t < kT; ++t) {
    // Async: pull the full 32x512 bf16 h-state (written by all WGs) into LDS via TDM,
    // overlapped with the W_ih half of this step's GEMM.
    if (wave == 0) tdm_load_2d(kLdsHprev, hstate, kB * kH, 1, kB * kH, 1, kB * kH);

    v8f acc0 = {}; v8f acc1 = {};

    // ---- GEMM 1: gates += h_in[:, t, :] @ W_ih^T   (A from global bf16 shadow) ----
    const bf16_t* a0p = arow0 + (size_t)t * kH;
    const bf16_t* a1p = arow1 + (size_t)t * kH;
#pragma unroll
    for (int kb = 0; kb < 16; ++kb) {
      const int ko = kb * 32 + kh * 16;
      v16bf a0 = cat8(*(const v8bf*)(a0p + ko), *(const v8bf*)(a0p + ko + 8));
      v16bf a1 = cat8(*(const v8bf*)(a1p + ko), *(const v8bf*)(a1p + ko + 8));
      v16bf bb = *(const v16bf*)(wih_s + (size_t)(lrow0 + r16) * kH + kb * 32 + kh * 16);
      acc0 = wmma_bf16(a0, bb, acc0);
      acc1 = wmma_bf16(a1, bb, acc1);
    }

    if (wave == 0) __builtin_amdgcn_s_wait_tensorcnt(0);
    __syncthreads();   // hprev_s now valid for all waves

    // ---- GEMM 2: gates += h_{t-1} @ W_hh^T   (A from LDS) ----
#pragma unroll
    for (int kb = 0; kb < 16; ++kb) {
      const int ko = kb * 32 + kh * 16;
      const bf16_t* h0 = hprev_s + (size_t)r16 * kH + ko;
      const bf16_t* h1 = hprev_s + (size_t)(16 + r16) * kH + ko;
      v16bf a0 = cat8(*(const v8bf*)h0, *(const v8bf*)(h0 + 8));
      v16bf a1 = cat8(*(const v8bf*)h1, *(const v8bf*)(h1 + 8));
      v16bf bb = *(const v16bf*)(whh_s + (size_t)(lrow0 + r16) * kH + kb * 32 + kh * 16);
      acc0 = wmma_bf16(a0, bb, acc0);
      acc1 = wmma_bf16(a1, bb, acc1);
    }

    // ---- stage gate pre-activations to LDS: gbuf[(gate*32+u)][batch] ----
#pragma unroll
    for (int i = 0; i < 8; ++i) {
      int m = kh * 8 + i;
      gbuf[(gate * kJPW + ublk + r16) * kB + m]      = acc0[i];
      gbuf[(gate * kJPW + ublk + r16) * kB + 16 + m] = acc1[i];
    }
    __syncthreads();

    // ---- elementwise LSTM cell update (i,f,g,o -> c,h) ----
    {
      int b  = tid >> 3;          // 0..31
      int u0 = (tid & 7) * 4;     // 0,4,...,28
      float4 hv;
      float* hvp = &hv.x;
#pragma unroll
      for (int q = 0; q < 4; ++q) {
        int u = u0 + q;
        float iv = sigmoidf_(gbuf[(0 * kJPW + u) * kB + b] + bias_s[0 * kJPW + u]);
        float fv = sigmoidf_(gbuf[(1 * kJPW + u) * kB + b] + bias_s[1 * kJPW + u]);
        float gv = tanhf(gbuf[(2 * kJPW + u) * kB + b] + bias_s[2 * kJPW + u]);
        float ov = sigmoidf_(gbuf[(3 * kJPW + u) * kB + b] + bias_s[3 * kJPW + u]);
        float c = fv * cstate[u * kB + b] + iv * gv;
        cstate[u * kB + b] = c;
        float h = ov * tanhf(c);
        hvp[q] = h;
        hstate[(size_t)b * kH + j0 + u] = (bf16_t)h;
      }
      *(float4*)(outb + ((size_t)b * kT + t) * kH + j0 + u0) = hv;
    }
    grid_sync(bar);   // publish h slices before anyone TDM-loads the next state
  }
}

__global__ __launch_bounds__(256)
void ln_kernel(const float* __restrict__ outb, float* __restrict__ hbuf,
               bf16_t* __restrict__ shadow, const float* __restrict__ lns,
               const float* __restrict__ lnb, int layer) {
  int row  = blockIdx.x * 8 + (threadIdx.x >> 5);   // over B*T rows, one wave each
  int lane = threadIdx.x & 31;
  const float* pa = outb + (size_t)row * kH;
  const float* pb = hbuf + (size_t)row * kH;
  float v[16];
  float s = 0.0f;
#pragma unroll
  for (int i = 0; i < 16; ++i) { int idx = lane + i * 32; v[i] = pa[idx] + pb[idx]; s += v[i]; }
#pragma unroll
  for (int o = 16; o > 0; o >>= 1) s += __shfl_xor(s, o, 32);
  float mu = s * (1.0f / kH);
  float var = 0.0f;
#pragma unroll
  for (int i = 0; i < 16; ++i) { float d = v[i] - mu; var += d * d; }
#pragma unroll
  for (int o = 16; o > 0; o >>= 1) var += __shfl_xor(var, o, 32);
  float rs = rsqrtf(var * (1.0f / kH) + kEps);
  float*  po = hbuf + (size_t)row * kH;
  bf16_t* ps = shadow + (size_t)row * kH;
#pragma unroll
  for (int i = 0; i < 16; ++i) {
    int idx = lane + i * 32;
    float y = (v[i] - mu) * rs * lns[layer * kH + idx] + lnb[layer * kH + idx];
    po[idx] = y;
    ps[idx] = (bf16_t)y;
  }
}

__global__ __launch_bounds__(256)
void out_kernel(const float* __restrict__ hbuf, const float* __restrict__ ow,
                const float* __restrict__ ob, float* __restrict__ y) {
  int row  = blockIdx.x * 8 + (threadIdx.x >> 5);
  int lane = threadIdx.x & 31;
  const float* p = hbuf + (size_t)row * kH;
  float s = 0.0f;
#pragma unroll
  for (int i = 0; i < 16; ++i) { int idx = lane + i * 32; s += p[idx] * ow[idx]; }
#pragma unroll
  for (int o = 16; o > 0; o >>= 1) s += __shfl_xor(s, o, 32);
  if (lane == 0) y[row] = s + ob[0];
}

// ---------------- launcher ----------------
extern "C" void kernel_launch(void* const* d_in, const int* in_sizes, int n_in,
                              void* d_out, int out_size, void* d_ws, size_t ws_size,
                              hipStream_t stream) {
  (void)in_sizes; (void)n_in; (void)out_size; (void)ws_size;
  const float* x   = (const float*)d_in[0];
  const float* ipw = (const float*)d_in[1];
  const float* ipb = (const float*)d_in[2];
  const float* wih = (const float*)d_in[3];
  const float* whh = (const float*)d_in[4];
  const float* bih = (const float*)d_in[5];
  const float* bhh = (const float*)d_in[6];
  const float* lns = (const float*)d_in[7];
  const float* lnb = (const float*)d_in[8];
  const float* ow  = (const float*)d_in[9];
  const float* ob  = (const float*)d_in[10];
  float* y = (float*)d_out;

  char* ws = (char*)d_ws;
  float*    hbuf   = (float*)(ws + kOffHbuf);
  float*    outb   = (float*)(ws + kOffOutbuf);
  bf16_t*   shadow = (bf16_t*)(ws + kOffShadow);
  bf16_t*   wihb   = (bf16_t*)(ws + kOffWihb);
  bf16_t*   whhb   = (bf16_t*)(ws + kOffWhhb);
  float*    xp     = (float*)(ws + kOffXp);
  bf16_t*   hst    = (bf16_t*)(ws + kOffHstate);
  unsigned* bar    = (unsigned*)(ws + kOffBar);

  (void)hipFuncSetAttribute((const void*)recur_kernel,
                            hipFuncAttributeMaxDynamicSharedMemorySize, (int)kLdsTotal);

  inproj_kernel<<<(kB * kH + 255) / 256, 256, 0, stream>>>(x, ipw, ipb, xp, bar);
  bcast_kernel<<<(unsigned)((size_t)kB * kT * kH / 256), 256, 0, stream>>>(xp, hbuf, shadow);
  wconv_kernel<<<4096, 256, 0, stream>>>(wih, whh, wihb, whhb);
  for (int l = 0; l < kL; ++l) {
    recur_kernel<<<kNWG, kNTH, kLdsTotal, stream>>>(wihb, whhb, bih, bhh,
                                                    shadow, outb, hst, bar, l);
    ln_kernel<<<(kB * kT) / 8, 256, 0, stream>>>(outb, hbuf, shadow, lns, lnb, l);
  }
  out_kernel<<<(kB * kT) / 8, 256, 0, stream>>>(hbuf, ow, ob, y);
}
